// InferenceVectorQuantizer_43404939493734
// MI455X (gfx1250) — matrix operand order repaired and checked by reading, not compiled, not run on hardware
//
#include <hip/hip_runtime.h>
#include <hip/hip_bf16.h>

typedef __attribute__((ext_vector_type(16))) _Float16 v16h;
typedef __attribute__((ext_vector_type(4)))  _Float16 v4h;
typedef __attribute__((ext_vector_type(8)))  float    v8f;

#define VQ_N 500000
#define VQ_K 512
#define VQ_D 64

// Per workgroup: 256 threads = 8 wave32; each wave handles 16 rows of X.
// LDS: f16-hi codebook [512][64], f16-lo codebook [512][64], e_sq f32 [512].
#define SMEM_HI_BYTES   (VQ_K * VQ_D * 2)          // 65536
#define SMEM_LO_BYTES   (VQ_K * VQ_D * 2)          // 65536
#define SMEM_ESQ_BYTES  (VQ_K * 4)                 // 2048
#define SMEM_TOTAL      (SMEM_HI_BYTES + SMEM_LO_BYTES + SMEM_ESQ_BYTES)

__device__ __forceinline__ v8f wmma_f16(v16h a, v16h b, v8f c) {
    return __builtin_amdgcn_wmma_f32_16x16x32_f16(
        /*neg_a=*/false, a, /*neg_b=*/false, b,
        /*c_mod=*/(short)0, c, /*reuse_a=*/false, /*reuse_b=*/false);
}

__global__ void __launch_bounds__(256)
vq_argmin_kernel(const float* __restrict__ X,      // [N][64]
                 const float* __restrict__ E,      // [512][64]
                 float* __restrict__ Q,            // [N][64] quantized out
                 int*   __restrict__ IDX,          // [N] argmin out
                 int n_rows) {
    extern __shared__ __align__(16) char smem[];
    _Float16* emb_hi = (_Float16*)smem;                                   // [512*64]
    _Float16* emb_lo = (_Float16*)(smem + SMEM_HI_BYTES);                 // [512*64]
    float*    esq    = (float*)   (smem + SMEM_HI_BYTES + SMEM_LO_BYTES); // [512]

    const int tid  = threadIdx.x;
    const int lane = tid & 31;
    const int wave = tid >> 5;
    const int half = (lane >> 4) & 1;   // 0: lanes 0-15, 1: lanes 16-31
    const int l16  = lane & 15;

    // ---- Stage codebook into LDS as f16 hi/lo split (hi + lo == f32 to ~2^-24) ----
    {
        const float4* E4 = (const float4*)E;
        for (int i4 = tid; i4 < (VQ_K * VQ_D) / 4; i4 += 256) {   // 32 b128 loads/thread
            float4 v = E4[i4];
            v4h hi, lo;
            hi[0] = (_Float16)v.x; lo[0] = (_Float16)(v.x - (float)hi[0]);
            hi[1] = (_Float16)v.y; lo[1] = (_Float16)(v.y - (float)hi[1]);
            hi[2] = (_Float16)v.z; lo[2] = (_Float16)(v.z - (float)hi[2]);
            hi[3] = (_Float16)v.w; lo[3] = (_Float16)(v.w - (float)hi[3]);
            *(v4h*)(emb_hi + (size_t)i4 * 4) = hi;   // 8B ds stores
            *(v4h*)(emb_lo + (size_t)i4 * 4) = lo;
        }
        // e_sq in f32 (4-way partial sums enable b128 loads)
        for (int c = tid; c < VQ_K; c += 256) {
            const float4* er = (const float4*)(E + (size_t)c * VQ_D);
            float4 s = {0.f, 0.f, 0.f, 0.f};
            #pragma unroll
            for (int k = 0; k < VQ_D / 4; ++k) {
                float4 v = er[k];
                s.x = __builtin_fmaf(v.x, v.x, s.x);
                s.y = __builtin_fmaf(v.y, v.y, s.y);
                s.z = __builtin_fmaf(v.z, v.z, s.z);
                s.w = __builtin_fmaf(v.w, v.w, s.w);
            }
            esq[c] = (s.x + s.y) + (s.z + s.w);
        }
    }
    __syncthreads();

    // ---- Build A fragments (16x32 f16 A-layout) for this wave's 16 rows ----
    // Lanes 0-15: elems 0..15 -> K koff..koff+15 (koff=kb*32)
    // Lanes 16-31: same with koff += 16.
    const long rowBase = (long)blockIdx.x * 128 + (long)wave * 16;
    long rlong = rowBase + l16;
    if (rlong >= n_rows) rlong = n_rows - 1;        // clamp tail (stores guarded below)
    const float* xr = X + (size_t)rlong * VQ_D;

    v16h Ahi[2], Alo[2];
    #pragma unroll
    for (int kb = 0; kb < 2; ++kb) {
        const int koff = kb * 32 + half * 16;
        #pragma unroll
        for (int j = 0; j < 8; ++j) {
            float v0 = xr[koff + j];
            _Float16 h0 = (_Float16)v0;
            Ahi[kb][j] = h0;
            Alo[kb][j] = (_Float16)(v0 - (float)h0);
            float v1 = xr[koff + 8 + j];
            _Float16 h1 = (_Float16)v1;
            Ahi[kb][8 + j] = h1;
            Alo[kb][8 + j] = (_Float16)(v1 - (float)h1);
        }
    }

    // ---- Sweep codebook in 16-column tiles; per-lane running argmin ----
    // score[m][n] = e_sq[n] - 2 * dot(x_m, e_n)   (x_sq constant under argmin_n)
    float best[8];
    int   bidx[8];
    #pragma unroll
    for (int v = 0; v < 8; ++v) { best[v] = 3.4e38f; bidx[v] = 0; }

    const int kB = half * 16;   // B-layout: lanes 0-15 hold K 0-15, lanes 16-31 hold K 16-31

    #pragma unroll 2
    for (int nt = 0; nt < VQ_K / 16; ++nt) {
        const int n = nt * 16 + l16;            // codebook column for this lane
        const _Float16* bh = emb_hi + (size_t)n * VQ_D + kB;
        const _Float16* bl = emb_lo + (size_t)n * VQ_D + kB;
        v16h Bhi0 = *(const v16h*)(bh);         // K 0..31 block
        v16h Bhi1 = *(const v16h*)(bh + 32);    // K 32..63 block
        v16h Blo0 = *(const v16h*)(bl);
        v16h Blo1 = *(const v16h*)(bl + 32);
        const float es = esq[n];

        // Three independent depth-2 accumulator chains -> hazard slots filled by
        // sibling WMMAs instead of v_nop (f16 WMMA RAW needs 1+4 co-exec slots).
        v8f acc0 = {};                          // xh . eh
        v8f acc1 = {};                          // xl . eh
        v8f acc2 = {};                          // xh . el
        acc0 = wmma_f16(Ahi[0], Bhi0, acc0);
        acc1 = wmma_f16(Alo[0], Bhi0, acc1);
        acc2 = wmma_f16(Ahi[0], Blo0, acc2);
        acc0 = wmma_f16(Ahi[1], Bhi1, acc0);
        acc1 = wmma_f16(Alo[1], Bhi1, acc1);
        acc2 = wmma_f16(Ahi[1], Blo1, acc2);

        #pragma unroll
        for (int v = 0; v < 8; ++v) {
            float dot = acc0[v] + (acc1[v] + acc2[v]);
            float score = __builtin_fmaf(-2.f, dot, es);
            if (score < best[v]) { best[v] = score; bidx[v] = n; }
        }
    }

    // ---- Cross-lane min-reduce within each 16-lane half (columns of D tile) ----
    // D layout: element v of lane l is row m = v + 8*half, column n ≡ (l&15) (mod 16).
    #pragma unroll
    for (int off = 1; off < 16; off <<= 1) {
        #pragma unroll
        for (int v = 0; v < 8; ++v) {
            float ob = __shfl_xor(best[v], off, 32);
            int   oi = __shfl_xor(bidx[v], off, 32);
            if (ob < best[v] || (ob == best[v] && oi < bidx[v])) {
                best[v] = ob; bidx[v] = oi;
            }
        }
    }

    // ---- Write indices + gather exact f32 code vectors from global embedding ----
    #pragma unroll
    for (int v = 0; v < 8; ++v) {
        const long row = rowBase + v + half * 8;
        if (row < n_rows) {
            const int idx = bidx[v];
            if (l16 == 0) IDX[row] = idx;
            // 16 lanes of this half write 4 floats each = full 64-float row
            const float4 q = *((const float4*)(E + (size_t)idx * VQ_D) + l16);
            *((float4*)(Q + (size_t)row * VQ_D) + l16) = q;
        }
    }
}

extern "C" void kernel_launch(void* const* d_in, const int* in_sizes, int n_in,
                              void* d_out, int out_size, void* d_ws, size_t ws_size,
                              hipStream_t stream) {
    (void)in_sizes; (void)n_in; (void)d_ws; (void)ws_size; (void)out_size;
    const float* flat_x    = (const float*)d_in[0];   // [N,64] f32
    const float* embedding = (const float*)d_in[1];   // [512,64] f32

    float* q_out = (float*)d_out;                           // [N,64] f32
    int*   i_out = (int*)(q_out + (size_t)VQ_N * VQ_D);     // [N] int32

    const int rows_per_block = 128;  // 8 waves x 16 rows
    const int blocks = (VQ_N + rows_per_block - 1) / rows_per_block;
    vq_argmin_kernel<<<blocks, 256, SMEM_TOTAL, stream>>>(
        flat_x, embedding, q_out, i_out, VQ_N);
}
